// MemoryAugmentedCNN_57208964383248
// MI455X (gfx1250) — matrix operand order repaired and checked by reading, not compiled, not run on hardware
//
#include <hip/hip_runtime.h>
#include <hip/hip_fp16.h>
#include <hip/hip_bf16.h>

typedef __attribute__((ext_vector_type(16))) _Float16 v16h;
typedef __attribute__((ext_vector_type(8)))  _Float16 v8h;
typedef __attribute__((ext_vector_type(8)))  float    v8f;

// ---------------------------------------------------------------------------
// Pack a f32 [K, N] matrix into f16 WMMA-B fragment layout with K/N padding:
//   dst[(kblk*Npad + n)*32 + (k - kblk*32)] = (k<K && n<N) ? src[k*N+n] : 0
// A lane's B fragment (kb, n, half) is then 16 contiguous halves (32B).
// ---------------------------------------------------------------------------
__global__ void pack_b_kernel(const float* __restrict__ src, _Float16* __restrict__ dst,
                              int K, int N, int Npad, int total) {
  int idx = blockIdx.x * blockDim.x + threadIdx.x;
  if (idx >= total) return;
  int e    = idx & 31;
  int n    = (idx >> 5) % Npad;
  int kblk = idx / (32 * Npad);
  int k    = kblk * 32 + e;
  float v  = (k < K && n < N) ? src[(long)k * N + n] : 0.f;
  dst[idx] = (_Float16)v;
}

// ---------------------------------------------------------------------------
// Pack conv weights w[OC=64][CIN][3][3] (f32) into WMMA-B fragment layout with
// implicit-GEMM K ordered as k = (ky*3+kx)*CIN + ic  (matches NHWC gathers):
//   dst[(kblk*64 + oc)*32 + (k%32)] = w[oc][ic][ky][kx]
// ---------------------------------------------------------------------------
__global__ void pack_convw_kernel(const float* __restrict__ src, _Float16* __restrict__ dst,
                                  int CIN, int total) {
  int idx = blockIdx.x * blockDim.x + threadIdx.x;
  if (idx >= total) return;
  int e    = idx & 31;
  int oc   = (idx >> 5) & 63;
  int kblk = idx >> 11;                 // /(32*64)
  int k    = kblk * 32 + e;             // (pos, ic) order
  int ic   = k % CIN;
  int pos  = k / CIN;
  int ky = pos / 3, kx = pos - ky * 3;
  dst[idx] = (_Float16)src[((oc * CIN + ic) * 3 + ky) * 3 + kx];
}

// ---------------------------------------------------------------------------
// conv1: direct fp32 conv (3->32, 3x3 VALID). K=27: too small/odd for WMMA.
// x: [B,3,14,14], w: [32,3,3,3], preact out: [B,32,12,12] (NCHW)
// ---------------------------------------------------------------------------
__global__ void conv1_kernel(const float* __restrict__ x, const float* __restrict__ w1,
                             const float* __restrict__ b1, float* __restrict__ pre, int total) {
  int idx = blockIdx.x * blockDim.x + threadIdx.x;
  if (idx >= total) return;
  int s  = idx % 144;
  int oc = (idx / 144) % 32;
  int b  = idx / (144 * 32);
  int oy = s / 12, ox = s % 12;
  const float* xb = x + (long)b * 3 * 196;
  const float* w  = w1 + oc * 27;
  float acc = b1[oc];
#pragma unroll
  for (int ic = 0; ic < 3; ++ic)
#pragma unroll
    for (int ky = 0; ky < 3; ++ky)
#pragma unroll
      for (int kx = 0; kx < 3; ++kx)
        acc += xb[ic * 196 + (oy + ky) * 14 + (ox + kx)] * w[ic * 9 + ky * 3 + kx];
  pre[idx] = acc;
}

// ---------------------------------------------------------------------------
// BatchNorm batch statistics over [B, C, S]: sums[c]=sum, sums[C+c]=sumsq.
// ---------------------------------------------------------------------------
__global__ void bn_stats_kernel(const float* __restrict__ pre, float* __restrict__ sums,
                                int C, int S, int Bn) {
  int c = blockIdx.x;
  int total = Bn * S;
  int per = (total + gridDim.y - 1) / gridDim.y;
  int start = blockIdx.y * per;
  int end = start + per; if (end > total) end = total;
  float s = 0.f, s2 = 0.f;
  for (int i = start + threadIdx.x; i < end; i += blockDim.x) {
    int b = i / S, sp = i - b * S;
    float v = pre[((long)b * C + c) * S + sp];
    s += v; s2 += v * v;
  }
  __shared__ float r1[256], r2[256];
  int t = threadIdx.x;
  r1[t] = s; r2[t] = s2;
  __syncthreads();
  for (int off = 128; off > 0; off >>= 1) {
    if (t < off) { r1[t] += r1[t + off]; r2[t] += r2[t + off]; }
    __syncthreads();
  }
  if (t == 0) { atomicAdd(&sums[c], r1[0]); atomicAdd(&sums[C + c], r2[0]); }
}

__global__ void bn_finalize_kernel(const float* __restrict__ sums, const float* __restrict__ g,
                                   const float* __restrict__ be, float* __restrict__ ss,
                                   int C, float inv_n) {
  int c = threadIdx.x;
  if (c >= C) return;
  float mu  = sums[c] * inv_n;
  float var = sums[C + c] * inv_n - mu * mu;
  float sc  = g[c] * rsqrtf(var + 1e-5f);
  ss[c] = sc;
  ss[C + c] = be[c] - mu * sc;
}

// ---------------------------------------------------------------------------
// y = exact_gelu(x*scale + shift) -> f16.
// nhwc=1: write [B,S,C] (conv input layout); nhwc=0: keep [B,C,S] flatten.
// ---------------------------------------------------------------------------
__global__ void bn_gelu_kernel(const float* __restrict__ pre, const float* __restrict__ ss,
                               _Float16* __restrict__ act, int C, int S, int total, int nhwc) {
  int i = blockIdx.x * blockDim.x + threadIdx.x;
  if (i >= total) return;
  int s = i % S;
  int c = (i / S) % C;
  int b = i / (S * C);
  float v = pre[i] * ss[c] + ss[C + c];
  float gl = 0.5f * v * (1.0f + erff(v * 0.70710678118654752f));
  long o = nhwc ? (((long)b * S + s) * C + c) : (long)i;
  act[o] = (_Float16)gl;
}

// ---------------------------------------------------------------------------
// A-fragment helper: lane half h holds k = 16h..16h+15 (contiguous, matches packed B).
// ---------------------------------------------------------------------------
__device__ __forceinline__ v16h load_a_frag(const _Float16* p, int half) {
  v8h a0 = *(const v8h*)(p + half * 16);
  v8h a1 = *(const v8h*)(p + half * 16 + 8);
  return __builtin_shufflevector(a0, a1, 0, 1, 2, 3, 4, 5, 6, 7,
                                 8, 9, 10, 11, 12, 13, 14, 15);
}

// ---------------------------------------------------------------------------
// Implicit-GEMM conv via WMMA f16, Cout=64, act NHWC [B,HIN,HIN,CIN] f16,
// weights packed (pos,ic K-order). All fragment loads are contiguous b128.
// Block = 128 threads (4 waves); wave -> 32(M) x 64(N) tile (2x4 accums),
// so each B fragment feeds two WMMAs.
// ---------------------------------------------------------------------------
template <int CIN, int HIN, int HOUT>
__global__ void __launch_bounds__(128)
conv_wmma_kernel(const _Float16* __restrict__ act, const _Float16* __restrict__ wp,
                 const float* __restrict__ bias, float* __restrict__ pre) {
  constexpr int KT = CIN * 9;
  constexpr int S2 = HOUT * HOUT;
  int lane = threadIdx.x & 31;
  int wave = threadIdx.x >> 5;
  int half = lane >> 4;
  int l16  = lane & 15;
  int mBase = (blockIdx.x * 4 + wave) * 32;   // two 16-row M tiles per wave

  const _Float16* ain[2];
#pragma unroll
  for (int t = 0; t < 2; ++t) {
    int mrow = mBase + t * 16 + l16;
    int b  = mrow / S2;
    int sp = mrow - b * S2;
    int oy = sp / HOUT, ox = sp - (sp / HOUT) * HOUT;
    ain[t] = act + ((long)(b * HIN + oy) * HIN + ox) * CIN;
  }

  v8f acc[2][4] = {};
#pragma unroll
  for (int kb = 0; kb < KT; kb += 32) {
    // k = (ky*3+kx)*CIN + ic : one contiguous channel run per position
    int pos = kb / CIN;
    int ic0 = kb - pos * CIN;
    int ky = pos / 3, kx = pos - ky * 3;
    int aoff = (ky * HIN + kx) * CIN + ic0;
    v16h a0 = load_a_frag(ain[0] + aoff, half);
    v16h a1 = load_a_frag(ain[1] + aoff, half);
    const _Float16* bbase = wp + (long)(kb >> 5) * (64 * 32) + half * 16;
#pragma unroll
    for (int nt = 0; nt < 4; ++nt) {
      int oc = nt * 16 + l16;
      v16h bf = *(const v16h*)(bbase + oc * 32);
      acc[0][nt] = __builtin_amdgcn_wmma_f32_16x16x32_f16(
          false, a0, false, bf, (short)0, acc[0][nt], false, false);
      acc[1][nt] = __builtin_amdgcn_wmma_f32_16x16x32_f16(
          false, a1, false, bf, (short)0, acc[1][nt], false, false);
    }
  }
#pragma unroll
  for (int t = 0; t < 2; ++t)
#pragma unroll
    for (int nt = 0; nt < 4; ++nt) {
      int oc = nt * 16 + l16;
      float bv = bias[oc];
#pragma unroll
      for (int r = 0; r < 8; ++r) {
        int m  = mBase + t * 16 + half * 8 + r;
        int b2 = m / S2;
        int s2 = m - b2 * S2;
        pre[((long)b2 * 64 + oc) * S2 + s2] = acc[t][nt][r] + bv;
      }
    }
}

// ---------------------------------------------------------------------------
// Generic WMMA GEMM on padded operands (K multiple of 32, B packed, Npad
// multiple of 16). C = A1*B1 (+ A2*B2) (+bias). Wave -> 32(M) x 64(N).
// ---------------------------------------------------------------------------
__device__ __forceinline__ void gemm_pair(const _Float16* __restrict__ A,
                                          const _Float16* __restrict__ Bp,
                                          int K, int Npad, int mBase, int nBase,
                                          int half, int l16, v8f acc[2][4]) {
  const _Float16* arow0 = A + (long)(mBase + l16) * K;
  const _Float16* arow1 = arow0 + (long)16 * K;
  for (int kb = 0; kb < K; kb += 32) {
    v16h a0 = load_a_frag(arow0 + kb, half);
    v16h a1 = load_a_frag(arow1 + kb, half);
    const _Float16* bbase = Bp + (long)(kb >> 5) * Npad * 32 + half * 16;
#pragma unroll
    for (int nt = 0; nt < 4; ++nt) {
      int n = nBase + nt * 16 + l16;
      v16h bf = *(const v16h*)(bbase + n * 32);
      acc[0][nt] = __builtin_amdgcn_wmma_f32_16x16x32_f16(
          false, a0, false, bf, (short)0, acc[0][nt], false, false);
      acc[1][nt] = __builtin_amdgcn_wmma_f32_16x16x32_f16(
          false, a1, false, bf, (short)0, acc[1][nt], false, false);
    }
  }
}

__global__ void __launch_bounds__(128)
gemm_wmma_kernel(const _Float16* __restrict__ A1, const _Float16* __restrict__ B1, int K1,
                 const _Float16* __restrict__ A2, const _Float16* __restrict__ B2, int K2,
                 const float* __restrict__ bias,
                 float* __restrict__ Cf, _Float16* __restrict__ Ch,
                 int N, int Npad, int ldc) {
  int lane = threadIdx.x & 31;
  int wave = threadIdx.x >> 5;
  int half = lane >> 4;
  int l16  = lane & 15;
  int mBase = (blockIdx.x * 4 + wave) * 32;
  int nBase = blockIdx.y * 64;

  v8f acc[2][4] = {};
  gemm_pair(A1, B1, K1, Npad, mBase, nBase, half, l16, acc);
  if (A2) gemm_pair(A2, B2, K2, Npad, mBase, nBase, half, l16, acc);

#pragma unroll
  for (int nt = 0; nt < 4; ++nt) {
    int n = nBase + nt * 16 + l16;
    if (n >= N) continue;
    float bv = bias ? bias[n] : 0.f;
#pragma unroll
    for (int t = 0; t < 2; ++t)
#pragma unroll
      for (int r = 0; r < 8; ++r) {
        int m = mBase + t * 16 + half * 8 + r;
        float v = acc[t][nt][r] + bv;
        if (Cf) Cf[(long)m * ldc + n] = v;
        else    Ch[(long)m * ldc + n] = (_Float16)v;
      }
  }
}

// ---------------------------------------------------------------------------
// Row softmax: keys [rows, ld=1024] f32 (N=1000 valid) -> f16, zero-padded
// to ld so the downstream GEMM needs no K guards.
// ---------------------------------------------------------------------------
__global__ void softmax_kernel(const float* __restrict__ keys, _Float16* __restrict__ rw,
                               int N, int ld) {
  int row = blockIdx.x;
  const float* kr = keys + (long)row * ld;
  __shared__ float red[256];
  int t = threadIdx.x;

  float mx = -3.4e38f;
  for (int i = t; i < N; i += 256) mx = fmaxf(mx, kr[i]);
  red[t] = mx; __syncthreads();
  for (int off = 128; off > 0; off >>= 1) {
    if (t < off) red[t] = fmaxf(red[t], red[t + off]);
    __syncthreads();
  }
  float m = red[0]; __syncthreads();

  float s = 0.f;
  for (int i = t; i < N; i += 256) s += expf(kr[i] - m);
  red[t] = s; __syncthreads();
  for (int off = 128; off > 0; off >>= 1) {
    if (t < off) red[t] += red[t + off];
    __syncthreads();
  }
  float inv = 1.f / red[0];

  for (int i = t; i < ld; i += 256) {
    float v = (i < N) ? expf(kr[i] - m) * inv : 0.f;
    rw[(long)row * ld + i] = (_Float16)v;
  }
}

// ---------------------------------------------------------------------------
// Host orchestration
// ---------------------------------------------------------------------------
extern "C" void kernel_launch(void* const* d_in, const int* in_sizes, int n_in,
                              void* d_out, int out_size, void* d_ws, size_t ws_size,
                              hipStream_t stream) {
  (void)in_sizes; (void)n_in; (void)out_size; (void)ws_size;
  const float* x   = (const float*)d_in[0];
  const float* w1  = (const float*)d_in[1];
  const float* b1  = (const float*)d_in[2];
  const float* g1  = (const float*)d_in[3];
  const float* be1 = (const float*)d_in[4];
  const float* w2  = (const float*)d_in[5];
  const float* b2  = (const float*)d_in[6];
  const float* g2  = (const float*)d_in[7];
  const float* be2 = (const float*)d_in[8];
  const float* w3  = (const float*)d_in[9];
  const float* b3  = (const float*)d_in[10];
  const float* g3  = (const float*)d_in[11];
  const float* be3 = (const float*)d_in[12];
  const float* Wh  = (const float*)d_in[13];
  const float* bh  = (const float*)d_in[14];
  const float* mem = (const float*)d_in[15];
  const float* Wo  = (const float*)d_in[16];
  const float* bo  = (const float*)d_in[17];
  float* out = (float*)d_out;
  char* ws = (char*)d_ws;

  const size_t MB = 1024ull * 1024ull;
  // Region plan (reused across stages):
  float*    pre13 = (float*)   (ws + 0);        // preact1 37.7MB | preact3 33.6MB
  _Float16* act1  = (_Float16*)(ws + 40 * MB);  // act1 NHWC 18.9MB | hidden f16 16.8MB
  float*    pre2  = (float*)   (ws + 60 * MB);  // preact2 52.4MB | keys[2048,1024] 8.4MB
  _Float16* rw    = (_Float16*)(ws + 113 * MB); // softmax weights f16 [2048,1024] 4.2MB
  _Float16* act2  = (_Float16*)(ws + 118 * MB); // act2 NHWC 26.2MB
  _Float16* readh = (_Float16*)(ws + 145 * MB); // read f16 [2048,512] 2.1MB
  _Float16* wf    = (_Float16*)(ws + 148 * MB); // packed f16 weights ~19MB
  float*    stats = (float*)   (ws + 167 * MB); // per-channel sum/sumsq
  float*    ssbuf = stats + 128;                // per-channel scale/shift
  float*    keys  = pre2;                       // keys reuse preact2 region

  _Float16* w2p  = wf;                          // [9  blk][64][32]  18432
  _Float16* w3p  = w2p + 18432;                 // [18 blk][64][32]  36864
  _Float16* Whp  = w3p + 36864;                 // packed [4096,1024] 4,194,304
  _Float16* memp = Whp + (long)4096 * 1024;     // packed [1024, 512]   524,288
  _Float16* Wop  = memp + (long)1024 * 512;     // packed [4608,1024] 4,718,592

  // --- weight packing into WMMA fragment layouts (with N/K zero padding) ---
  pack_convw_kernel<<<(18432 + 255) / 256, 256, 0, stream>>>(w2, w2p, 32, 18432);
  pack_convw_kernel<<<(36864 + 255) / 256, 256, 0, stream>>>(w3, w3p, 64, 36864);
  {
    int t1 = 4096 * 1024;
    pack_b_kernel<<<(t1 + 255) / 256, 256, 0, stream>>>(Wh, Whp, 4096, 1000, 1024, t1);
    int t2 = 1024 * 512;
    pack_b_kernel<<<(t2 + 255) / 256, 256, 0, stream>>>(mem, memp, 1000, 512, 512, t2);
    int t3 = 4608 * 1024;
    pack_b_kernel<<<(t3 + 255) / 256, 256, 0, stream>>>(Wo, Wop, 4608, 1000, 1024, t3);
  }

  // --- stage 1: conv1 (fp32 direct) + BN + GELU -> act1 NHWC f16 ---
  const int T1 = 2048 * 32 * 144;
  conv1_kernel<<<(T1 + 255) / 256, 256, 0, stream>>>(x, w1, b1, pre13, T1);
  hipMemsetAsync(stats, 0, 128 * sizeof(float), stream);
  bn_stats_kernel<<<dim3(32, 64), 256, 0, stream>>>(pre13, stats, 32, 144, 2048);
  bn_finalize_kernel<<<1, 64, 0, stream>>>(stats, g1, be1, ssbuf, 32, 1.f / (2048.f * 144.f));
  bn_gelu_kernel<<<(T1 + 255) / 256, 256, 0, stream>>>(pre13, ssbuf, act1, 32, 144, T1, 1);

  // --- stage 2: conv2 WMMA (M=204800, N=64, K=288) ---
  conv_wmma_kernel<32, 12, 10><<<204800 / 128, 128, 0, stream>>>(act1, w2p, b2, pre2);
  const int T2 = 2048 * 64 * 100;
  hipMemsetAsync(stats, 0, 128 * sizeof(float), stream);
  bn_stats_kernel<<<dim3(64, 64), 256, 0, stream>>>(pre2, stats, 64, 100, 2048);
  bn_finalize_kernel<<<1, 64, 0, stream>>>(stats, g2, be2, ssbuf, 64, 1.f / (2048.f * 100.f));
  bn_gelu_kernel<<<(T2 + 255) / 256, 256, 0, stream>>>(pre2, ssbuf, act2, 64, 100, T2, 1);

  // --- stage 3: conv3 WMMA (M=131072, N=64, K=576) ---
  conv_wmma_kernel<64, 10, 8><<<131072 / 128, 128, 0, stream>>>(act2, w3p, b3, pre13);
  const int T3 = 2048 * 64 * 64;
  hipMemsetAsync(stats, 0, 128 * sizeof(float), stream);
  bn_stats_kernel<<<dim3(64, 64), 256, 0, stream>>>(pre13, stats, 64, 64, 2048);
  bn_finalize_kernel<<<1, 64, 0, stream>>>(stats, g3, be3, ssbuf, 64, 1.f / (2048.f * 64.f));
  bn_gelu_kernel<<<(T3 + 255) / 256, 256, 0, stream>>>(pre13, ssbuf, act1, 64, 64, T3, 0);
  // act1 now holds hidden f16 [2048, 4096] (NCHW flatten = reference reshape)

  // --- head GEMM: keys = hidden @ W_head + b_head (2048 x 4096 x 1000) ---
  gemm_wmma_kernel<<<dim3(2048 / 128, 16), 128, 0, stream>>>(
      act1, Whp, 4096, nullptr, nullptr, 0, bh, keys, nullptr, 1000, 1024, 1024);

  // --- softmax over classes -> read weights f16 (zero-padded to 1024) ---
  softmax_kernel<<<2048, 256, 0, stream>>>(keys, rw, 1000, 1024);

  // --- read GEMM: read = rw @ memory (2048 x 1024p x 512), f16 out ---
  gemm_wmma_kernel<<<dim3(2048 / 128, 8), 128, 0, stream>>>(
      rw, memp, 1024, nullptr, nullptr, 0, nullptr, nullptr, readh, 512, 512, 512);

  // --- out GEMM: out = hidden@Wo[0:4096] + read@Wo[4096:4608] + b_out ---
  gemm_wmma_kernel<<<dim3(2048 / 128, 16), 128, 0, stream>>>(
      act1, Wop, 4096, readh, Wop + (long)128 * 1024 * 32, 512, bo, out, nullptr,
      1000, 1024, 1000);
}
